// MPNNLayer_54443005444257
// MI455X (gfx1250) — compile-verified, matrix operand order checked by reading, not run on hardware
//
#include <hip/hip_runtime.h>

typedef __attribute__((ext_vector_type(16))) _Float16 v16h;
typedef __attribute__((ext_vector_type(8)))  float    v8f;
typedef __attribute__((ext_vector_type(4)))  float    v4f;

#define NODE_DIM 32
#define EDGE_DIM 16
#define HIDDEN   32
#define NCHUNK   33   // 32 chunks from w2 + 1 chunk from b2 (h==1)

// ---------------------------------------------------------------------------
// Kernel 0: permute w2 (f32 [32,1024]) + b2 (f32 [1024]) into f16 B-fragment
// layout:  bperm[((kc*2+t)*32 + lane)*16 + i]
//   n  = (lane&15) + 16*t,  hi = lane>>4,  d = i + 8*hi + (i>=8 ? 8 : 0)
//   kc<32 : w2[kc*1024 + d*32 + n]        kc==32 : b2[d*32 + n]
// so each lane of the WMMA wave loads its v16h B operand contiguously.
// ---------------------------------------------------------------------------
__global__ void k_prep_bperm(const float* __restrict__ w2,
                             const float* __restrict__ b2,
                             _Float16* __restrict__ bperm) {
  int tid = blockIdx.x * blockDim.x + threadIdx.x;
  if (tid >= NCHUNK * 1024) return;
  int i  = tid & 15;
  int L  = (tid >> 4) & 31;
  int t  = (tid >> 9) & 1;
  int kc = tid >> 10;
  int n  = (L & 15) + 16 * t;
  int hi = L >> 4;
  int d  = i + 8 * hi + (i >= 8 ? 8 : 0);
  float v = (kc < 32) ? w2[kc * 1024 + d * 32 + n] : b2[d * 32 + n];
  bperm[tid] = (_Float16)v;
}

// ---------------------------------------------------------------------------
// Kernel 1: zero the aggregation + count scratch (ws is poisoned by harness)
// ---------------------------------------------------------------------------
__global__ void k_zero(float* __restrict__ p, long n) {
  long i = (long)blockIdx.x * blockDim.x + threadIdx.x;
  if (i < n) p[i] = 0.0f;
}

// ---------------------------------------------------------------------------
// Kernel 2: edge MLP  h = relu(edge_attr @ w1 + b1)  -> f16, plus degree count
// ---------------------------------------------------------------------------
__global__ void k_edge_h(const float* __restrict__ ea,
                         const float* __restrict__ w1,
                         const float* __restrict__ b1,
                         const int*   __restrict__ eidx,   // [2,E] flat
                         _Float16* __restrict__ hws,
                         float* __restrict__ cnt,
                         int E) {
  __shared__ float w1s[EDGE_DIM * HIDDEN];
  __shared__ float b1s[HIDDEN];
  for (int i = threadIdx.x; i < EDGE_DIM * HIDDEN; i += blockDim.x) w1s[i] = w1[i];
  if (threadIdx.x < HIDDEN) b1s[threadIdx.x] = b1[threadIdx.x];
  __syncthreads();

  int e = blockIdx.x * blockDim.x + threadIdx.x;
  if (e >= E) return;

  float a[EDGE_DIM];
  const v4f* eap = (const v4f*)(ea + (size_t)e * EDGE_DIM);
#pragma unroll
  for (int j = 0; j < 4; ++j) {
    v4f v = eap[j];
    a[4 * j + 0] = v.x; a[4 * j + 1] = v.y; a[4 * j + 2] = v.z; a[4 * j + 3] = v.w;
  }
  float acc[HIDDEN];
#pragma unroll
  for (int k = 0; k < HIDDEN; ++k) acc[k] = b1s[k];
#pragma unroll
  for (int d = 0; d < EDGE_DIM; ++d) {
    float av = a[d];
#pragma unroll
    for (int k = 0; k < HIDDEN; ++k) acc[k] = fmaf(av, w1s[d * 32 + k], acc[k]);
  }
  v16h h0, h1;
#pragma unroll
  for (int k = 0; k < 16; ++k) {
    h0[k] = (_Float16)fmaxf(acc[k], 0.0f);
    h1[k] = (_Float16)fmaxf(acc[16 + k], 0.0f);
  }
  v16h* hp = (v16h*)(hws + (size_t)e * 32);
  hp[0] = h0;
  hp[1] = h1;

  // degree count for mean aggregation
  atomicAdd(&cnt[eidx[E + e]], 1.0f);
}

// ---------------------------------------------------------------------------
// Kernel 3 (hot): per-edge message GEMM via WMMA f32_16x16x32_f16.
// One wave = 32 edges (two M-tiles sharing the B fragments), so each K-chunk
// issues 4 independent WMMAs: they fill each other's WMMA->VALU hazard slots
// and halve the B-load traffic per edge.
// kc-th 32-wide K-chunk of Z is  h[e][kc] * x_src[e][0..31]  (kc==32 -> h=1).
// D fragments are scattered straight into agg[] with global f32 atomics.
// ---------------------------------------------------------------------------
__global__ void __launch_bounds__(128)
k_msg(const float* __restrict__ x,
      const int*   __restrict__ eidx,
      const _Float16* __restrict__ hws,
      const _Float16* __restrict__ bperm,
      float* __restrict__ agg,
      int E) {
  const int lane = threadIdx.x & 31;
  const int wave = threadIdx.x >> 5;
  const long tile = (long)blockIdx.x * 4 + wave;
  const long base_e = tile * 32;
  if (base_e >= E) return;           // whole-wave exit only (EXEC stays all-1s)

  const int m  = lane & 15;
  const int hi = lane >> 4;
  long e0 = base_e + m;        if (e0 >= E) e0 = (long)E - 1;  // M-tile 0
  long e1 = base_e + 16 + m;   if (e1 >= E) e1 = (long)E - 1;  // M-tile 1

  // --- A sources: gather x rows (16 f16 per lane, documented A layout) ---
  const int src0 = eidx[e0];
  const int src1 = eidx[e1];
  const float* xp0 = x + (size_t)src0 * 32;
  const float* xp1 = x + (size_t)src1 * 32;
  v4f a00 = *(const v4f*)(xp0 + 8 * hi);
  v4f a01 = *(const v4f*)(xp0 + 8 * hi + 4);
  v4f a02 = *(const v4f*)(xp0 + 16 + 8 * hi);
  v4f a03 = *(const v4f*)(xp0 + 20 + 8 * hi);
  v4f a10 = *(const v4f*)(xp1 + 8 * hi);
  v4f a11 = *(const v4f*)(xp1 + 8 * hi + 4);
  v4f a12 = *(const v4f*)(xp1 + 16 + 8 * hi);
  v4f a13 = *(const v4f*)(xp1 + 20 + 8 * hi);
  v16h xh0, xh1;
#pragma unroll
  for (int j = 0; j < 4; ++j) {
    xh0[j]      = (_Float16)a00[j];
    xh0[4 + j]  = (_Float16)a01[j];
    xh0[8 + j]  = (_Float16)a02[j];
    xh0[12 + j] = (_Float16)a03[j];
    xh1[j]      = (_Float16)a10[j];
    xh1[4 + j]  = (_Float16)a11[j];
    xh1[8 + j]  = (_Float16)a12[j];
    xh1[12 + j] = (_Float16)a13[j];
  }

  // --- per-edge h rows (32 f16 each) kept in registers; the unrolled loop
  //     turns element extraction into compile-time subvector reads ---
  const v16h* hp0 = (const v16h*)(hws + (size_t)e0 * 32);
  const v16h* hp1 = (const v16h*)(hws + (size_t)e1 * 32);
  v16h hlo0 = hp0[0], hhi0 = hp0[1];
  v16h hlo1 = hp1[0], hhi1 = hp1[1];

  v8f c00 = {};   // M-tile 0, N = 0..15
  v8f c01 = {};   // M-tile 0, N = 16..31
  v8f c10 = {};   // M-tile 1, N = 0..15
  v8f c11 = {};   // M-tile 1, N = 16..31

#pragma unroll
  for (int kc = 0; kc < NCHUNK; ++kc) {
    _Float16 hv0 = (kc < 16) ? hlo0[kc]
                 : (kc < 32) ? hhi0[kc - 16]
                             : (_Float16)1.0f;           // b2 chunk
    _Float16 hv1 = (kc < 16) ? hlo1[kc]
                 : (kc < 32) ? hhi1[kc - 16]
                             : (_Float16)1.0f;
    v16h a0 = xh0 * hv0;                                 // v_pk_mul_f16 x8
    v16h a1 = xh1 * hv1;
    v16h bA = *(const v16h*)(bperm + ((size_t)(kc * 2 + 0) * 32 + lane) * 16);
    v16h bB = *(const v16h*)(bperm + ((size_t)(kc * 2 + 1) * 32 + lane) * 16);
    c00 = __builtin_amdgcn_wmma_f32_16x16x32_f16(false, a0, false, bA, (short)0, c00, false, false);
    c10 = __builtin_amdgcn_wmma_f32_16x16x32_f16(false, a1, false, bA, (short)0, c10, false, false);
    c01 = __builtin_amdgcn_wmma_f32_16x16x32_f16(false, a0, false, bB, (short)0, c01, false, false);
    c11 = __builtin_amdgcn_wmma_f32_16x16x32_f16(false, a1, false, bB, (short)0, c11, false, false);
  }

  // --- scatter: D element r of lane = msg[edge base + r + 8*hi][N] ---
  int dsts0[8], dsts1[8];
  bool ok0[8], ok1[8];
#pragma unroll
  for (int r = 0; r < 8; ++r) {
    long er0 = base_e + r + 8 * hi;
    long er1 = base_e + 16 + r + 8 * hi;
    ok0[r] = (er0 < E);
    ok1[r] = (er1 < E);
    dsts0[r] = ok0[r] ? eidx[E + er0] : 0;
    dsts1[r] = ok1[r] ? eidx[E + er1] : 0;
  }
  const int n0 = lane & 15;
#pragma unroll
  for (int r = 0; r < 8; ++r) {
    if (ok0[r]) {
      atomicAdd(&agg[(size_t)dsts0[r] * 32 + n0],      c00[r]);
      atomicAdd(&agg[(size_t)dsts0[r] * 32 + 16 + n0], c01[r]);
    }
    if (ok1[r]) {
      atomicAdd(&agg[(size_t)dsts1[r] * 32 + n0],      c10[r]);
      atomicAdd(&agg[(size_t)dsts1[r] * 32 + 16 + n0], c11[r]);
    }
  }
}

// ---------------------------------------------------------------------------
// Kernel 4: per-node finalize.
// messages = agg/max(cnt,1) + x@root + bias_conv
// out      = relu([x, messages] @ w3 + b3)
// ---------------------------------------------------------------------------
__global__ void __launch_bounds__(128)
k_final(const float* __restrict__ x,
        const float* __restrict__ root,
        const float* __restrict__ bconv,
        const float* __restrict__ w3,
        const float* __restrict__ b3,
        const float* __restrict__ agg,
        const float* __restrict__ cnt,
        float* __restrict__ out,
        int N) {
  __shared__ float roots[32 * 32];
  __shared__ float w3s[64 * 32];
  __shared__ float bcs[32];
  __shared__ float b3s[32];
  for (int i = threadIdx.x; i < 32 * 32; i += blockDim.x) roots[i] = root[i];
  for (int i = threadIdx.x; i < 64 * 32; i += blockDim.x) w3s[i] = w3[i];
  if (threadIdx.x < 32) { bcs[threadIdx.x] = bconv[threadIdx.x]; b3s[threadIdx.x] = b3[threadIdx.x]; }
  __syncthreads();

  int n = blockIdx.x * blockDim.x + threadIdx.x;
  if (n >= N) return;

  float xr[32];
  const v4f* xp = (const v4f*)(x + (size_t)n * 32);
#pragma unroll
  for (int j = 0; j < 8; ++j) {
    v4f v = xp[j];
    xr[4 * j + 0] = v.x; xr[4 * j + 1] = v.y; xr[4 * j + 2] = v.z; xr[4 * j + 3] = v.w;
  }
  float inv = 1.0f / fmaxf(cnt[n], 1.0f);

  float msg[32];
#pragma unroll
  for (int o = 0; o < 32; ++o) msg[o] = agg[(size_t)n * 32 + o] * inv + bcs[o];
#pragma unroll
  for (int d = 0; d < 32; ++d) {
    float xv = xr[d];
#pragma unroll
    for (int o = 0; o < 32; ++o) msg[o] = fmaf(xv, roots[d * 32 + o], msg[o]);
  }

  float acc[32];
#pragma unroll
  for (int o = 0; o < 32; ++o) acc[o] = b3s[o];
#pragma unroll
  for (int d = 0; d < 32; ++d) {
    float xv = xr[d];
#pragma unroll
    for (int o = 0; o < 32; ++o) acc[o] = fmaf(xv, w3s[d * 32 + o], acc[o]);
  }
#pragma unroll
  for (int d = 0; d < 32; ++d) {
    float mv = msg[d];
#pragma unroll
    for (int o = 0; o < 32; ++o) acc[o] = fmaf(mv, w3s[(32 + d) * 32 + o], acc[o]);
  }
#pragma unroll
  for (int o = 0; o < 32; ++o) out[(size_t)n * 32 + o] = fmaxf(acc[o], 0.0f);
}

// ---------------------------------------------------------------------------
extern "C" void kernel_launch(void* const* d_in, const int* in_sizes, int n_in,
                              void* d_out, int out_size, void* d_ws, size_t ws_size,
                              hipStream_t stream) {
  const float* x     = (const float*)d_in[0];
  const int*   eidx  = (const int*)d_in[1];     // [2,E]
  const float* ea    = (const float*)d_in[2];   // [E,16]
  const float* w1    = (const float*)d_in[3];
  const float* b1    = (const float*)d_in[4];
  const float* w2    = (const float*)d_in[5];
  const float* b2    = (const float*)d_in[6];
  const float* root  = (const float*)d_in[7];
  const float* bconv = (const float*)d_in[8];
  const float* w3    = (const float*)d_in[9];
  const float* b3    = (const float*)d_in[10];
  float* out = (float*)d_out;

  const int N = in_sizes[0] / NODE_DIM;
  const int E = in_sizes[2] / EDGE_DIM;

  // workspace layout
  char* ws = (char*)d_ws;
  _Float16* bperm = (_Float16*)ws;                               // 33*1024*2 = 67,584 B
  _Float16* hws   = (_Float16*)(ws + 131072);                    // E*32 f16  = 16 MB
  float*    agg   = (float*)(ws + 131072 + (size_t)E * 64);      // N*32 f32
  float*    cnt   = agg + (size_t)N * 32;                        // N f32

  // 0) permute w2/b2 into f16 B-fragment layout
  k_prep_bperm<<<(NCHUNK * 1024 + 255) / 256, 256, 0, stream>>>(w2, b2, bperm);

  // 1) zero agg + cnt (contiguous: N*32 + N floats)
  long zn = (long)N * 33;
  k_zero<<<(int)((zn + 255) / 256), 256, 0, stream>>>(agg, zn);

  // 2) edge MLP -> h (f16), plus degree counts
  k_edge_h<<<(E + 255) / 256, 256, 0, stream>>>(ea, w1, b1, eidx, hws, cnt, E);

  // 3) hot WMMA kernel: 1 wave per 32 edges, 4 waves per block
  k_msg<<<(E + 127) / 128, 128, 0, stream>>>(x, eidx, hws, bperm, agg, E);

  // 4) node finalize
  k_final<<<(N + 127) / 128, 128, 0, stream>>>(x, root, bconv, w3, b3, agg, cnt, out, N);
}
